// Attention_472446403100
// MI455X (gfx1250) — matrix operand order repaired and checked by reading, not compiled, not run on hardware
//
#include <hip/hip_runtime.h>

typedef __attribute__((ext_vector_type(16))) __bf16 v16bf;
typedef __attribute__((ext_vector_type(8)))  __bf16 v8bf;
typedef __attribute__((ext_vector_type(4)))  __bf16 v4bf;
typedef __attribute__((ext_vector_type(8)))  float  v8f;

constexpr int Bsz = 2, Sseq = 2048, Dmodel = 1024, Hn = 16, DKd = 64;
constexpr int Mrows = Bsz * Sseq;   // 4096
constexpr float NEG_MAX = -3.402823466e38f;

// ---------------------------------------------------------------------------
// bf16-WMMA GEMM: Out[m,n] = sum_k X[m,k] * W[n,k] + bias[n]
//   XBF16: input element type (f32 converted in-kernel, or bf16 pass-through)
//   MODE : 0 = f32 [M,N] output
//          1 = head-split bf16 [B,H,S,DK]
//          2 = head-split TRANSPOSED bf16 [B,H,DK,S]   (for V^T)
// Block = 256 thr (8 waves), tile M=32 x N=64, K-step 32.
// ---------------------------------------------------------------------------
template <bool XBF16, int MODE>
__global__ void gemm_wmma_kernel(const void* __restrict__ Xraw,
                                 const float* __restrict__ W,
                                 const float* __restrict__ bias,
                                 void* __restrict__ Outraw,
                                 int N, int K)
{
    __shared__ __align__(16) __bf16 sX[32][32];
    __shared__ __align__(16) __bf16 sW[64][32];

    const float*  Xf = (const float*)Xraw;
    const __bf16* Xb = (const __bf16*)Xraw;

    const int tid  = threadIdx.x;       // 0..255
    const int lane = tid & 31;
    const int wv   = tid >> 5;          // 0..7
    const int m0   = blockIdx.x * 32;
    const int n0   = blockIdx.y * 64;

    v8f acc = {};
    for (int k0 = 0; k0 < K; k0 += 32) {
        // ---- stage X tile 32x32: one float4 (or v4bf) per thread ----
        {
            int row = tid >> 3, kq = (tid & 7) * 4;
            size_t gi = (size_t)(m0 + row) * K + k0 + kq;
            v4bf pk;
            if constexpr (XBF16) {
                pk = *(const v4bf*)(Xb + gi);
            } else {
                float4 f = *(const float4*)(Xf + gi);
                pk[0] = (__bf16)f.x; pk[1] = (__bf16)f.y;
                pk[2] = (__bf16)f.z; pk[3] = (__bf16)f.w;
                if (k0 + 32 < K) __builtin_prefetch(Xf + gi + 32, 0, 1);
            }
            *(v4bf*)&sX[row][kq] = pk;
        }
        // ---- stage W tile 64x32: two float4 per thread ----
        #pragma unroll
        for (int i = 0; i < 2; ++i) {
            int e = i * 256 + tid;
            int row = e >> 3, kq = (e & 7) * 4;
            size_t gi = (size_t)(n0 + row) * K + k0 + kq;
            float4 f = *(const float4*)(W + gi);
            v4bf pk;
            pk[0] = (__bf16)f.x; pk[1] = (__bf16)f.y;
            pk[2] = (__bf16)f.z; pk[3] = (__bf16)f.w;
            *(v4bf*)&sW[row][kq] = pk;
            if (k0 + 32 < K) __builtin_prefetch(W + gi + 32, 0, 1);
        }
        __syncthreads();

        // ---- A fragment (16x32 bf16, ISA 7.12.2): two contiguous 8-elem runs ----
        const int mrow  = ((wv & 4) << 2) + (lane & 15);   // waves 0-3 rows 0-15, 4-7 rows 16-31
        const int kbase = (lane >> 4) << 4;
        v8bf x0 = *(const v8bf*)&sX[mrow][kbase];
        v8bf x1 = *(const v8bf*)&sX[mrow][kbase + 8];
        v16bf a = __builtin_shufflevector(x0, x1, 0,1,2,3,4,5,6,7,8,9,10,11,12,13,14,15);

        // ---- B fragment (32x16 bf16): element (k,n) = W[n,k]; contiguous 16 ----
        const int nB  = ((wv & 3) << 4) + (lane & 15);
        const int kbB = (lane >> 4) << 4;
        v16bf b = *(const v16bf*)&sW[nB][kbB];

        acc = __builtin_amdgcn_wmma_f32_16x16x32_bf16(
            false, a, false, b, (short)0, acc, false, false);
        __syncthreads();
    }

    // ---- epilogue ----
    const int nC = n0 + ((wv & 3) << 4) + (lane & 15);
    const int mW = m0 + ((wv & 4) << 2);
    const float bc = bias[nC];
    #pragma unroll
    for (int r = 0; r < 8; ++r) {
        int m = mW + r + ((lane >> 4) << 3);
        float v = acc[r] + bc;
        if constexpr (MODE == 0) {
            ((float*)Outraw)[(size_t)m * N + nC] = v;
        } else {
            int bb = m / Sseq, ss = m % Sseq;
            int hh = nC >> 6, dk = nC & 63;
            if constexpr (MODE == 1)
                ((__bf16*)Outraw)[(((size_t)bb * Hn + hh) * Sseq + ss) * DKd + dk] = (__bf16)v;
            else // MODE == 2: V^T  [B,H,DK,S]
                ((__bf16*)Outraw)[(((size_t)bb * Hn + hh) * DKd + dk) * Sseq + ss] = (__bf16)v;
        }
    }
}

// ---------------------------------------------------------------------------
// Fused attention: one block = 16 query rows of one (b,h).
// QK^T (WMMA bf16) -> mask/scale -> row softmax in 128KB dynamic LDS
// -> write attn weights f32 once -> fused PV (WMMA bf16, V^T layout) -> ctx bf16.
// Block = 256 thr (8 waves).
// ---------------------------------------------------------------------------
__global__ void attn_wmma_kernel(const __bf16* __restrict__ Qw,   // [B,H,S,DK]
                                 const __bf16* __restrict__ Kw,   // [B,H,S,DK]
                                 const __bf16* __restrict__ Vt,   // [B,H,DK,S]
                                 const float* __restrict__ mask,  // [B,1,S,S]
                                 float* __restrict__ attn_out,    // [B,H,S,S]
                                 __bf16* __restrict__ ctx_ws)     // [B,S,D]
{
    extern __shared__ char smemraw[];
    float* scores = (float*)smemraw;          // 16*2048
    float* red    = scores + 16 * 2048;       // 16*16
    float* rowmax = red + 256;                // 16
    float* rowsum = rowmax + 16;              // 16
    float* ctxp   = rowsum + 16;              // 2*16*64

    const int tid  = threadIdx.x;
    const int lane = tid & 31;
    const int wv   = tid >> 5;                // 0..7
    const int qt   = blockIdx.x;              // 0..127
    const int bh   = blockIdx.y;              // 0..31
    const int bb   = bh >> 4;
    const size_t base = (size_t)bh * Sseq * DKd;

    // Q fragments (DK=64 -> two K=32 fragments); contiguous v8bf global loads
    v16bf a0, a1;
    {
        const int m     = lane & 15;
        const int kbase = (lane >> 4) << 4;
        const __bf16* qp = Qw + base + (size_t)(qt * 16 + m) * DKd;
        v8bf q00 = *(const v8bf*)(qp + kbase);
        v8bf q01 = *(const v8bf*)(qp + kbase + 8);
        v8bf q10 = *(const v8bf*)(qp + kbase + 32);
        v8bf q11 = *(const v8bf*)(qp + kbase + 40);
        a0 = __builtin_shufflevector(q00, q01, 0,1,2,3,4,5,6,7,8,9,10,11,12,13,14,15);
        a1 = __builtin_shufflevector(q10, q11, 0,1,2,3,4,5,6,7,8,9,10,11,12,13,14,15);
    }

    // ---- scores: 128 key tiles striped over 8 waves ----
    for (int j = 0; j < 16; ++j) {
        const int nt  = wv + 8 * j;
        const int kbB = (lane >> 4) << 4;
        const __bf16* kp = Kw + base + (size_t)(nt * 16 + (lane & 15)) * DKd;
        v16bf b0 = *(const v16bf*)(kp + kbB);
        v16bf b1 = *(const v16bf*)(kp + kbB + 32);
        v8f acc = {};
        acc = __builtin_amdgcn_wmma_f32_16x16x32_bf16(false, a0, false, b0, (short)0, acc, false, false);
        acc = __builtin_amdgcn_wmma_f32_16x16x32_bf16(false, a1, false, b1, (short)0, acc, false, false);
        #pragma unroll
        for (int r = 0; r < 8; ++r) {
            int m    = r + ((lane >> 4) << 3);
            int col  = nt * 16 + (lane & 15);
            int qrow = qt * 16 + m;
            float s  = acc[r] * 0.125f;   // 1/sqrt(64)
            float mk = mask[((size_t)bb * Sseq + qrow) * Sseq + col];
            s += (1.0f - mk) * NEG_MAX;
            scores[m * 2048 + col] = s;
        }
    }
    __syncthreads();

    // ---- softmax: 16 threads per row, 16-stride (bank-conflict-free) ----
    const int row = tid >> 4, sub = tid & 15;
    float mx = NEG_MAX;
    for (int i = 0; i < 128; ++i)
        mx = fmaxf(mx, scores[row * 2048 + sub + 16 * i]);
    red[row * 16 + sub] = mx;
    __syncthreads();
    if (sub == 0) {
        float mm = red[row * 16];
        for (int j = 1; j < 16; ++j) mm = fmaxf(mm, red[row * 16 + j]);
        rowmax[row] = mm;
    }
    __syncthreads();
    const float rm = rowmax[row];
    float sum = 0.f;
    for (int i = 0; i < 128; ++i) {
        int c = sub + 16 * i;
        float e = __expf(scores[row * 2048 + c] - rm);
        scores[row * 2048 + c] = e;
        sum += e;
    }
    red[row * 16 + sub] = sum;
    __syncthreads();
    if (sub == 0) {
        float ss = 0.f;
        for (int j = 0; j < 16; ++j) ss += red[row * 16 + j];
        rowsum[row] = ss;
    }
    __syncthreads();
    const float inv = 1.0f / rowsum[row];
    {
        int qrow = qt * 16 + row;
        float* arow = attn_out + ((size_t)bh * Sseq + qrow) * Sseq;
        for (int i = 0; i < 128; ++i) {
            int c = sub + 16 * i;
            float p = scores[row * 2048 + c] * inv;
            scores[row * 2048 + c] = p;
            arow[c] = p;               // materialize attention_weights once
        }
    }
    __syncthreads();

    // ---- fused PV: ctx[16,64] = P[16,2048] @ V[2048,64], V in [DK,S] (V^T) ----
    {
        const int nt  = wv & 3;           // N subtile of DK
        const int kh  = wv >> 2;          // split-K half
        const int mA  = lane & 15;
        const int kbA = (lane >> 4) << 4;
        const int nB  = nt * 16 + (lane & 15);            // dk index
        const int kbB = (lane >> 4) << 4;
        const __bf16* vtp = Vt + (size_t)bh * DKd * Sseq + (size_t)nB * Sseq;
        v8f acc = {};
        for (int k0 = kh * 1024; k0 < kh * 1024 + 1024; k0 += 32) {
            // A from LDS probs (f32 -> bf16), two contiguous 8-float runs
            const float* sp = &scores[mA * 2048 + k0];
            float4 f0 = *(const float4*)(sp + kbA);
            float4 f1 = *(const float4*)(sp + kbA + 4);
            float4 f2 = *(const float4*)(sp + kbA + 8);
            float4 f3 = *(const float4*)(sp + kbA + 12);
            v16bf a;
            a[0]=(__bf16)f0.x; a[1]=(__bf16)f0.y; a[2]=(__bf16)f0.z; a[3]=(__bf16)f0.w;
            a[4]=(__bf16)f1.x; a[5]=(__bf16)f1.y; a[6]=(__bf16)f1.z; a[7]=(__bf16)f1.w;
            a[8]=(__bf16)f2.x; a[9]=(__bf16)f2.y; a[10]=(__bf16)f2.z; a[11]=(__bf16)f2.w;
            a[12]=(__bf16)f3.x; a[13]=(__bf16)f3.y; a[14]=(__bf16)f3.z; a[15]=(__bf16)f3.w;
            // B: element (k,n) = V^T[n][k] -> contiguous 16 bf16 in S
            v16bf b = *(const v16bf*)(vtp + k0 + kbB);
            acc = __builtin_amdgcn_wmma_f32_16x16x32_bf16(
                false, a, false, b, (short)0, acc, false, false);
        }
        #pragma unroll
        for (int r = 0; r < 8; ++r) {
            int m = r + ((lane >> 4) << 3);
            ctxp[(kh * 16 + m) * 64 + nt * 16 + (lane & 15)] = acc[r];
        }
    }
    __syncthreads();

    // split-K reduce + store ctx bf16 in [B,S,D] concat-head layout
    const int hh = bh & 15;
    #pragma unroll
    for (int i = 0; i < 4; ++i) {
        int e = i * 256 + tid;
        int m = e >> 6, n = e & 63;
        float v = ctxp[m * 64 + n] + ctxp[(16 + m) * 64 + n];
        int qrow = qt * 16 + m;
        ctx_ws[((size_t)bb * Sseq + qrow) * Dmodel + hh * DKd + n] = (__bf16)v;
    }
}

// ---------------------------------------------------------------------------
extern "C" void kernel_launch(void* const* d_in, const int* in_sizes, int n_in,
                              void* d_out, int out_size, void* d_ws, size_t ws_size,
                              hipStream_t stream)
{
    (void)in_sizes; (void)n_in; (void)out_size; (void)ws_size;
    const float* query = (const float*)d_in[0];
    const float* key_  = (const float*)d_in[1];
    const float* value = (const float*)d_in[2];
    const float* mask  = (const float*)d_in[3];
    const float* wq = (const float*)d_in[4];
    const float* bq = (const float*)d_in[5];
    const float* wk = (const float*)d_in[6];
    const float* bk = (const float*)d_in[7];
    const float* wv = (const float*)d_in[8];
    const float* bv = (const float*)d_in[9];
    const float* wo = (const float*)d_in[10];
    const float* bo = (const float*)d_in[11];

    float* out  = (float*)d_out;                         // [B,S,D]
    float* attn = out + (size_t)Bsz * Sseq * Dmodel;     // [B,H,S,S]

    // workspace: Q,K head-split + V^T + ctx, all bf16 (32 MB total)
    const size_t hsElems = (size_t)Bsz * Hn * Sseq * DKd;   // 4M
    char* ws = (char*)d_ws;
    void* q_ws = (void*)(ws);
    void* k_ws = (void*)(ws + 2 * hsElems);
    void* v_ws = (void*)(ws + 4 * hsElems);
    void* c_ws = (void*)(ws + 6 * hsElems);

    dim3 gg(Mrows / 32, Dmodel / 64);   // 128 x 16
    gemm_wmma_kernel<false, 1><<<gg, 256, 0, stream>>>(query, wq, bq, q_ws, Dmodel, Dmodel);
    gemm_wmma_kernel<false, 1><<<gg, 256, 0, stream>>>(key_,  wk, bk, k_ws, Dmodel, Dmodel);
    gemm_wmma_kernel<false, 2><<<gg, 256, 0, stream>>>(value, wv, bv, v_ws, Dmodel, Dmodel);

    const size_t lds = (size_t)(16 * 2048 + 256 + 16 + 16 + 2 * 16 * 64) * sizeof(float); // ~137 KB (<320 KB WGP LDS)
    attn_wmma_kernel<<<dim3(Sseq / 16, Bsz * Hn), 256, lds, stream>>>(
        (const __bf16*)q_ws, (const __bf16*)k_ws, (const __bf16*)v_ws, mask, attn, c_ws ? (__bf16*)c_ws : nullptr);

    gemm_wmma_kernel<true, 0><<<gg, 256, 0, stream>>>(c_ws, wo, bo, out, Dmodel, Dmodel);
}
